// LSTM_56925496541780
// MI455X (gfx1250) — matrix operand order repaired and checked by reading, not compiled, not run on hardware
//
#include <hip/hip_runtime.h>
#include <hip/hip_bf16.h>

// ---------------------------------------------------------------------------
// 3-layer bidirectional LSTM, B=64 T=256 D=1024 H=512, OUT=10.
//   1) convert weights to bf16 once; fold bih+bhh.
//   2) per layer: WMMA GEMM precomputes Gin = seq @ Wih^T + b, stored in
//      WMMA C-fragment layout (f32) so the scan reloads it vectorized.
//   3) per layer: persistent recurrent kernel (1 block/direction, 32 waves),
//      gates = Gin_t + h @ Whh^T via v_wmma_f32_16x16x32_bf16, h in LDS,
//      c in registers, h written back with global_store_async_from_lds_b128.
//   4) tiny final projection.
// Simple (non-pipelined) K-loops: occupancy hides load latency; manual
// double-buffering caused scratch spills (round-2 lesson).
// ---------------------------------------------------------------------------

typedef __attribute__((ext_vector_type(16))) __bf16 v16bf;
typedef __attribute__((ext_vector_type(8)))  __bf16 v8bf;
typedef __attribute__((ext_vector_type(8)))  float  v8f;

#define B_    64
#define T_    256
#define D_    1024
#define H_    512
#define G4_   2048          // 4*H
#define M_    (T_*B_)       // 16384
#define NT_   (G4_/16)      // 128 n-tiles
#define FRAG_ (32*8)        // floats per 16x16 f32 fragment

union frag16 { v16bf v; v8bf h[2]; };

// A fragment, 16x32 bf16, row-major source, leading dim lda (elements).
// lanes 0-15: K=0..7 & 16..23 ; lanes 16-31: K=8..15 & 24..31.
__device__ __forceinline__ v16bf load_a_frag(const __bf16* a, int lda, int lane) {
    const int m  = lane & 15;
    const int hi = lane >> 4;
    const __bf16* p = a + (size_t)m * lda + 8 * hi;
    frag16 f;
    f.h[0] = *(const v8bf*)(p);
    f.h[1] = *(const v8bf*)(p + 16);
    return f.v;
}

// B fragment, 32x16 bf16 with B[k][n] = W[n][k], W row-major [N,K], ld = ldk.
// lanes 0-15: K=0..7 & 16..23 ; lanes 16-31: K=8..15 & 24..31.
__device__ __forceinline__ v16bf load_bT_frag(const __bf16* w, int ldk, int lane) {
    const int n  = lane & 15;
    const int hi = lane >> 4;
    const __bf16* p = w + (size_t)n * ldk + 8 * hi;
    frag16 f;
    f.h[0] = *(const v8bf*)(p);
    f.h[1] = *(const v8bf*)(p + 16);
    return f.v;
}

__device__ __forceinline__ v8f wmma_bf16(v16bf a, v16bf b, v8f c) {
    return __builtin_amdgcn_wmma_f32_16x16x32_bf16(
        false, a, false, b, (short)0, c, false, false);
}

// sigmoid: one v_exp_f32 + v_rcp_f32
__device__ __forceinline__ float sigmoidf_(float x) {
    return __frcp_rn(1.0f + __expf(-x));
}
// tanh(x) = 2*sigmoid(2x) - 1 : one v_exp_f32 + v_rcp_f32 + fma
__device__ __forceinline__ float tanhf_(float x) {
    return __builtin_fmaf(2.0f, sigmoidf_(2.0f * x), -1.0f);
}

// Gin fragment offset: [mtile][ntile][lane][8] f32
__device__ __forceinline__ size_t gin_off(int mtile, int ntile, int lane) {
    return ((size_t)mtile * NT_ + ntile) * FRAG_ + (size_t)lane * 8;
}

// CDNA5 async LDS->global store (ASYNCcnt); 16 bytes per op.
__device__ __forceinline__ void async_store_b128(const __bf16* gdst, unsigned lds_off) {
    unsigned long long ga = (unsigned long long)(uintptr_t)gdst;
    asm volatile("global_store_async_from_lds_b128 %0, %1, off"
                 :: "v"(ga), "v"(lds_off) : "memory");
}
__device__ __forceinline__ void wait_asynccnt0() {
    asm volatile("s_wait_asynccnt 0x0" ::: "memory");
}

// --------------------------- weight conversion -----------------------------
__global__ void lstm_convert_weights(const float* __restrict__ wih,
                                     const float* __restrict__ whh,
                                     const float* __restrict__ bih,
                                     const float* __restrict__ bhh,
                                     __bf16* __restrict__ wih_bf,
                                     __bf16* __restrict__ whh_bf,
                                     float*  __restrict__ bias) {
    const size_t n_wih = (size_t)3 * 2 * G4_ * D_;
    const size_t n_whh = (size_t)3 * 2 * G4_ * H_;
    const size_t n_b   = (size_t)3 * 2 * G4_;
    size_t i      = (size_t)blockIdx.x * blockDim.x + threadIdx.x;
    size_t stride = (size_t)gridDim.x * blockDim.x;
    for (size_t k = i; k < n_wih; k += stride) wih_bf[k] = (__bf16)wih[k];
    for (size_t k = i; k < n_whh; k += stride) whh_bf[k] = (__bf16)whh[k];
    for (size_t k = i; k < n_b;   k += stride) bias[k]   = bih[k] + bhh[k];
}

// ------------------- x: [B,T,D] f32 -> seq: [T,B,D] bf16 -------------------
__global__ void lstm_convert_x(const float* __restrict__ x,
                               __bf16* __restrict__ seq) {
    size_t i = (size_t)blockIdx.x * blockDim.x + threadIdx.x;
    if (i >= (size_t)B_ * T_ * D_) return;
    int d     = (int)(i & (D_ - 1));
    size_t bt = i >> 10;
    int t = (int)(bt & (T_ - 1));
    int b = (int)(bt >> 8);
    seq[((size_t)t * B_ + b) * D_ + d] = (__bf16)x[i];
}

// ------------- Gin[dir] = seq @ Wih[dir]^T + bias[dir]  (fragment f32) -----
// grid: (NT_/4, M_/128, 2), block 256 (8 waves, each a 16x64 tile).
__global__ void lstm_input_gemm(const __bf16* __restrict__ seq,
                                const __bf16* __restrict__ wih,
                                const float*  __restrict__ bias,
                                float* __restrict__ gin) {
    const int lane = threadIdx.x & 31;
    const int wave = threadIdx.x >> 5;
    const int dir  = blockIdx.z;
    const int n0   = blockIdx.x * 64;
    const int m0   = blockIdx.y * 128 + wave * 16;

    const __bf16* w  = wih + (size_t)dir * G4_ * D_;
    const float*  bb = bias + (size_t)dir * G4_;
    float* g = gin + (size_t)dir * (M_/16) * NT_ * FRAG_;

    const int n = lane & 15;

    v8f acc[4];
    for (int j = 0; j < 4; ++j) {
        float bv = bb[n0 + j * 16 + n];
        for (int r = 0; r < 8; ++r) acc[j][r] = bv;
    }

    const __bf16* abase = seq + (size_t)m0 * D_;

    for (int k0 = 0; k0 < D_; k0 += 32) {
        v16bf af = load_a_frag(abase + k0, D_, lane);
        #pragma unroll
        for (int j = 0; j < 4; ++j) {
            v16bf bfr = load_bT_frag(w + (size_t)(n0 + j * 16) * D_ + k0, D_, lane);
            acc[j] = wmma_bf16(af, bfr, acc[j]);
        }
    }

    const int mtile = m0 >> 4;
    #pragma unroll
    for (int j = 0; j < 4; ++j)
        *(v8f*)(g + gin_off(mtile, (n0 >> 4) + j, lane)) = acc[j];
}

// ----------------------------- recurrent scan ------------------------------
// grid: 2 blocks (dir), block 1024 = 32 waves. Wave w owns hidden cols
// [16w,16w+16): gate tiles {16w, +512, +1024, +1536} x 4 M-tiles.
__global__ __launch_bounds__(1024)
void lstm_recurrent(const float*  __restrict__ gin,   // fragment layout
                    const __bf16* __restrict__ whh,   // [2][G4_][H_]
                    __bf16* __restrict__ seq_out) {   // [T_*B_][2H]
    __shared__ __bf16 hbuf[B_ * H_];

    const int lane = threadIdx.x & 31;
    const int wave = threadIdx.x >> 5;
    const int dir  = blockIdx.x;
    const int j0   = wave * 16;
    const int n    = lane & 15;
    const int hi   = lane >> 4;

    const float*  g = gin + (size_t)dir * (M_/16) * NT_ * FRAG_;
    const __bf16* w = whh + (size_t)dir * G4_ * H_;

    for (int i = threadIdx.x; i < B_ * H_; i += 1024) hbuf[i] = (__bf16)0.0f;
    __syncthreads();

    float creg[4][8];
    for (int mt = 0; mt < 4; ++mt)
        for (int r = 0; r < 8; ++r) creg[mt][r] = 0.0f;

    for (int step = 0; step < T_; ++step) {
        const int t  = dir ? (T_ - 1 - step) : step;
        const int tn = dir ? (t > 0 ? t - 1 : t) : (t < T_ - 1 ? t + 1 : t);

        // vectorized fragment loads of precomputed input projections
        v8f acc[4][4];
        #pragma unroll
        for (int mt = 0; mt < 4; ++mt)
            #pragma unroll
            for (int gt = 0; gt < 4; ++gt)
                acc[mt][gt] = *(const v8f*)(g + gin_off(t * 4 + mt, gt * 32 + wave, lane));

        // prefetch next step's fragments into cache (global_prefetch_b8)
        #pragma unroll
        for (int mt = 0; mt < 4; ++mt)
            #pragma unroll
            for (int gt = 0; gt < 4; ++gt)
                __builtin_prefetch(g + gin_off(tn * 4 + mt, gt * 32 + wave, lane), 0, 3);

        // gates += h_prev @ Whh^T
        for (int k0 = 0; k0 < H_; k0 += 32) {
            v16bf bfr[4];
            #pragma unroll
            for (int gt = 0; gt < 4; ++gt)
                bfr[gt] = load_bT_frag(w + (size_t)(gt * H_ + j0) * H_ + k0, H_, lane);
            #pragma unroll
            for (int mt = 0; mt < 4; ++mt) {
                v16bf af = load_a_frag(hbuf + (size_t)mt * 16 * H_ + k0, H_, lane);
                #pragma unroll
                for (int gt = 0; gt < 4; ++gt)
                    acc[mt][gt] = wmma_bf16(af, bfr[gt], acc[mt][gt]);
            }
        }

        // own async stores (prev step) must finish reading hbuf before rewrite
        wait_asynccnt0();
        __syncthreads();   // + everyone's LDS reads / async issues done

        // LSTM cell elementwise; write h to LDS only
        #pragma unroll
        for (int mt = 0; mt < 4; ++mt) {
            #pragma unroll
            for (int r = 0; r < 8; ++r) {
                const float iv = acc[mt][0][r];
                const float fv = acc[mt][1][r];
                const float gv = acc[mt][2][r];
                const float ov = acc[mt][3][r];
                float c = creg[mt][r];
                c = sigmoidf_(fv) * c + sigmoidf_(iv) * tanhf_(gv);
                creg[mt][r] = c;
                const float hv = sigmoidf_(ov) * tanhf_(c);
                const int row = mt * 16 + 8 * hi + r;
                hbuf[row * H_ + j0 + n] = (__bf16)hv;
            }
        }
        __syncthreads();   // h complete in LDS

        // async h write-back LDS->global; overlaps next step's K-loop.
        {
            const int chunk = threadIdx.x;      // 64B per chunk, 1024 chunks
            const int row   = chunk >> 4;
            const int seg   = chunk & 15;
            const unsigned lds_off = (unsigned)((row * H_ + seg * 32) * 2);
            const __bf16* gbase =
                seq_out + ((size_t)t * B_ + row) * (2 * H_) + dir * H_ + seg * 32;
            #pragma unroll
            for (int q = 0; q < 4; ++q)
                async_store_b128(gbase + q * 8, lds_off + q * 16);
        }
        // (no barrier needed here: next step only reads hbuf)
    }
    // s_endpgm waits all counters (incl. ASYNCcnt) idle.
}

// --------------------------- final projection ------------------------------
__global__ void lstm_final_proj(const __bf16* __restrict__ last,  // [B_][2H]
                                const float* __restrict__ wout,   // [10][2H]
                                const float* __restrict__ bout,
                                float* __restrict__ out) {
    int i = blockIdx.x * blockDim.x + threadIdx.x;
    if (i >= B_ * 10) return;
    int b = i / 10, o = i % 10;
    float s = bout[o];
    const __bf16* h = last + (size_t)b * (2 * H_);
    const float*  wr = wout + (size_t)o * (2 * H_);
    for (int k = 0; k < 2 * H_; ++k) s += (float)h[k] * wr[k];
    out[i] = s;
}

// ---------------------------------------------------------------------------
extern "C" void kernel_launch(void* const* d_in, const int* in_sizes, int n_in,
                              void* d_out, int out_size, void* d_ws, size_t ws_size,
                              hipStream_t stream) {
    const float* x    = (const float*)d_in[0];
    const float* Wih  = (const float*)d_in[1];
    const float* Whh  = (const float*)d_in[2];
    const float* bih  = (const float*)d_in[3];
    const float* bhh  = (const float*)d_in[4];
    const float* Wout = (const float*)d_in[5];
    const float* bout = (const float*)d_in[6];
    float* out = (float*)d_out;

    char* ws = (char*)d_ws;
    size_t off = 0;
    auto take = [&](size_t bytes) -> void* {
        void* p = ws + off;
        off = (off + bytes + 255) & ~(size_t)255;
        return p;
    };
    __bf16* wih_bf = (__bf16*)take((size_t)3 * 2 * G4_ * D_ * sizeof(__bf16));
    __bf16* whh_bf = (__bf16*)take((size_t)3 * 2 * G4_ * H_ * sizeof(__bf16));
    float*  bias   = (float*) take((size_t)3 * 2 * G4_ * sizeof(float));
    __bf16* seqA   = (__bf16*)take((size_t)M_ * D_ * sizeof(__bf16));
    __bf16* seqB   = (__bf16*)take((size_t)M_ * D_ * sizeof(__bf16));
    float*  gin    = (float*) take((size_t)2 * M_ * G4_ * sizeof(float));

    hipLaunchKernelGGL(lstm_convert_weights, dim3(2048), dim3(256), 0, stream,
                       Wih, Whh, bih, bhh, wih_bf, whh_bf, bias);
    hipLaunchKernelGGL(lstm_convert_x, dim3((B_ * T_ * D_) / 256), dim3(256), 0, stream,
                       x, seqA);

    for (int l = 0; l < 3; ++l) {
        const __bf16* seq_in  = (l & 1) ? seqB : seqA;
        __bf16*       seq_out = (l & 1) ? seqA : seqB;
        hipLaunchKernelGGL(lstm_input_gemm, dim3(NT_ / 4, M_ / 128, 2), dim3(256), 0, stream,
                           seq_in,
                           wih_bf + (size_t)l * 2 * G4_ * D_,
                           bias + (size_t)l * 2 * G4_,
                           gin);
        hipLaunchKernelGGL(lstm_recurrent, dim3(2), dim3(1024), 0, stream,
                           gin,
                           whh_bf + (size_t)l * 2 * G4_ * H_,
                           seq_out);
    }

    const __bf16* last = seqB + (size_t)(T_ - 1) * B_ * (2 * H_);
    hipLaunchKernelGGL(lstm_final_proj, dim3((B_ * 10 + 127) / 128), dim3(128), 0, stream,
                       last, Wout, bout, out);
}
